// DeformableTransformerDecoderLayer_876173328771
// MI455X (gfx1250) — compile-verified, matrix operand order checked by reading, not run on hardware
//
#include <hip/hip_runtime.h>
#include <math.h>

// ---------------- problem constants (fixed by reference) ----------------
#define T_  900
#define B_  16
#define D_  256
#define H_  8
#define DH_ 32
#define L_  4
#define P_  4
#define DFF_ 1024
#define LV_ 13294
#define M1_ (T_ * B_)          // 14400 activation rows
#define MV_ (LV_ * B_)         // 212704 memory/value rows
#define TP_ 928                // T padded to multiple of 32 for b128 V loads

typedef __attribute__((ext_vector_type(16))) __bf16 v16bf;
typedef __attribute__((ext_vector_type(8)))  float  v8f;

union BF { v16bf v; unsigned int u[8]; uint4 q[2]; unsigned short s[16]; };

__device__ __forceinline__ unsigned short f2bf(float f) {
  union { float f; unsigned int u; } x; x.f = f;
  unsigned int u = x.u;
  return (unsigned short)((u + 0x7FFFu + ((u >> 16) & 1u)) >> 16); // RNE
}
__device__ __forceinline__ float bf2f(unsigned short h) {
  union { unsigned int u; float f; } x; x.u = ((unsigned int)h) << 16;
  return x.f;
}
__device__ __forceinline__ v8f wmma_bf16(v16bf a, v16bf b, v8f c) {
  return __builtin_amdgcn_wmma_f32_16x16x32_bf16(false, a, false, b, (short)0, c,
                                                 false, false);
}

// ------ WMMA GEMM: C(M,N) = A(M,K)bf16 x Wt(N,K)bf16 + bias ------------------
// All operands bf16 in memory: every fragment = two b128 loads, zero conversion.
// Wave tile 32(M) x 64(N); block = 8 waves (4Mx2N) -> 128 x 128 tile.
__global__ __launch_bounds__(256) void k_gemm(
    const unsigned short* __restrict__ A, int lda,     // (M,K) bf16
    const unsigned short* __restrict__ Wt,             // (N,K) bf16
    const float* __restrict__ bias,
    float* __restrict__ Cf, unsigned short* __restrict__ Cb, int ldc,
    int M, int N, int K, int relu) {
  const int lane = threadIdx.x & 31;
  const int w    = threadIdx.x >> 5;
  const int m0   = blockIdx.x * 128 + (w >> 1) * 32;
  const int n0   = blockIdx.y * 128 + (w & 1) * 64;
  const int mr   = lane & 15;
  const int g    = lane >> 4;

  v8f acc[2][4];
#pragma unroll
  for (int i = 0; i < 2; ++i)
#pragma unroll
    for (int j = 0; j < 4; ++j) acc[i][j] = (v8f){};

  // OOB rows: clamp to row 0 (always valid); stores are guarded.
  const unsigned short* Ap[2];
  {
    int r0 = m0 + mr;      Ap[0] = A + (size_t)(r0 < M ? r0 : 0) * lda;
    int r1 = m0 + 16 + mr; Ap[1] = A + (size_t)(r1 < M ? r1 : 0) * lda;
  }
  const unsigned short* Bp[4];
#pragma unroll
  for (int j = 0; j < 4; ++j) {
    int n = n0 + j * 16 + mr;
    Bp[j] = Wt + (size_t)(n < N ? n : 0) * K;
  }

  for (int kk = 0; kk < K; kk += 32) {
    const int kb = kk + 8 * g;
    BF a[2], b[4];
#pragma unroll
    for (int i = 0; i < 2; ++i) {
      a[i].q[0] = *(const uint4*)(Ap[i] + kb);
      a[i].q[1] = *(const uint4*)(Ap[i] + kb + 16);
    }
#pragma unroll
    for (int j = 0; j < 4; ++j) {
      b[j].q[0] = *(const uint4*)(Bp[j] + kb);
      b[j].q[1] = *(const uint4*)(Bp[j] + kb + 16);
    }
#pragma unroll
    for (int i = 0; i < 2; ++i)
#pragma unroll
      for (int j = 0; j < 4; ++j)
        acc[i][j] = wmma_bf16(a[i].v, b[j].v, acc[i][j]);
  }

#pragma unroll
  for (int i = 0; i < 2; ++i) {
    const int rb = m0 + i * 16 + (g ? 8 : 0);
#pragma unroll
    for (int j = 0; j < 4; ++j) {
      const int col = n0 + j * 16 + (lane & 15);
      if (col >= N) continue;
      const float bv = bias ? bias[col] : 0.f;
#pragma unroll
      for (int r = 0; r < 8; ++r) {
        const int row = rb + r;
        if (row < M) {
          float val = acc[i][j][r] + bv;
          if (relu) val = fmaxf(val, 0.f);
          if (Cf) Cf[(size_t)row * ldc + col] = val;
          else    Cb[(size_t)row * ldc + col] = f2bf(val);
        }
      }
    }
  }
}

// ------------- flash attention, one wave per 16-row Q tile -------------------
// qh/kh: (T*B, D) bf16.  vtb: (B*D, TP_) bf16 (keys contiguous per channel).
__global__ __launch_bounds__(256) void k_attn(
    const unsigned short* __restrict__ qh, const unsigned short* __restrict__ kh,
    const unsigned short* __restrict__ vtb, unsigned short* __restrict__ sa) {
  __shared__ unsigned short pshare[8][16 * 32];
  const int lane = threadIdx.x & 31;
  const int w    = threadIdx.x >> 5;
  const int bh   = blockIdx.x;            // B*H = 128
  const int b    = bh >> 3, h = bh & 7;
  const int qt   = blockIdx.y * 8 + w;
  const int numQT = (T_ + 15) / 16;       // 57
  if (qt >= numQT) return;
  const int mr = lane & 15, g = lane >> 4;
  const int cbase = h * DH_;
  const int kb = 8 * g;
  const float qs = 0.17677669529663687f;  // 1/sqrt(32)

  BF aq;
  {
    int t = qt * 16 + mr;
    const unsigned short* Qp = qh + ((size_t)(t < T_ ? t : 0) * B_ + b) * D_ + cbase;
    aq.q[0] = *(const uint4*)(Qp + kb);
    aq.q[1] = *(const uint4*)(Qp + kb + 16);
  }
  const unsigned short* Vp0 = vtb + (size_t)(b * D_ + cbase + mr) * TP_;
  const unsigned short* Vp1 = Vp0 + (size_t)16 * TP_;

  float mrow[8], lrow[8];
  v8f o0 = {}, o1 = {};
#pragma unroll
  for (int r = 0; r < 8; ++r) { mrow[r] = -3.0e38f; lrow[r] = 0.f; }

  unsigned short* pl = pshare[w];
  const int nChunk = (T_ + 31) / 32;      // 29
  for (int ch = 0; ch < nChunk; ++ch) {
    const int kbase = ch * 32;
    BF bk0, bk1;
    {
      int key0 = kbase + mr, key1 = kbase + 16 + mr;
      if (key0 >= T_) key0 = 0;           // clamp; masked via scores below
      if (key1 >= T_) key1 = 0;
      const unsigned short* K0 = kh + ((size_t)key0 * B_ + b) * D_ + cbase;
      const unsigned short* K1 = kh + ((size_t)key1 * B_ + b) * D_ + cbase;
      bk0.q[0] = *(const uint4*)(K0 + kb);
      bk0.q[1] = *(const uint4*)(K0 + kb + 16);
      bk1.q[0] = *(const uint4*)(K1 + kb);
      bk1.q[1] = *(const uint4*)(K1 + kb + 16);
    }
    v8f s0 = {}, s1 = {};
    s0 = wmma_bf16(aq.v, bk0.v, s0);
    s1 = wmma_bf16(aq.v, bk1.v, s1);

    const int c0 = kbase + (lane & 15), c1 = c0 + 16;
    float p0[8], p1[8];
#pragma unroll
    for (int r = 0; r < 8; ++r) {
      float x0 = s0[r] * qs, x1 = s1[r] * qs;
      if (c0 >= T_) x0 = -3.0e38f;
      if (c1 >= T_) x1 = -3.0e38f;
      float loc = fmaxf(x0, x1);
#pragma unroll
      for (int sft = 1; sft < 16; sft <<= 1)
        loc = fmaxf(loc, __shfl_xor(loc, sft, 16));
      float mnew  = fmaxf(mrow[r], loc);
      float alpha = __expf(mrow[r] - mnew);
      mrow[r] = mnew;
      float e0 = __expf(x0 - mnew), e1 = __expf(x1 - mnew);
      p0[r] = e0; p1[r] = e1;
      float ssum = e0 + e1;
#pragma unroll
      for (int sft = 1; sft < 16; sft <<= 1)
        ssum += __shfl_xor(ssum, sft, 16);
      lrow[r] = lrow[r] * alpha + ssum;
      o0[r] *= alpha; o1[r] *= alpha;
    }
    // transpose P (C-layout -> A-layout) through per-wave LDS
#pragma unroll
    for (int r = 0; r < 8; ++r) {
      int row = r + (g ? 8 : 0);
      pl[row * 32 + (lane & 15)]      = f2bf(p0[r]);
      pl[row * 32 + 16 + (lane & 15)] = f2bf(p1[r]);
    }
    BF ap;
    ap.q[0] = *(const uint4*)(pl + mr * 32 + kb);
    ap.q[1] = *(const uint4*)(pl + mr * 32 + 16 + kb);
    BF bv0, bv1;
    bv0.q[0] = *(const uint4*)(Vp0 + kbase + kb);
    bv0.q[1] = *(const uint4*)(Vp0 + kbase + 16 + kb);
    bv1.q[0] = *(const uint4*)(Vp1 + kbase + kb);
    bv1.q[1] = *(const uint4*)(Vp1 + kbase + 16 + kb);
    o0 = wmma_bf16(ap.v, bv0.v, o0);
    o1 = wmma_bf16(ap.v, bv1.v, o1);
  }
#pragma unroll
  for (int r = 0; r < 8; ++r) {
    int t = qt * 16 + r + (g ? 8 : 0);
    if (t < T_) {
      float inv = 1.0f / lrow[r];
      size_t base = ((size_t)t * B_ + b) * D_ + cbase;
      sa[base + (lane & 15)]      = f2bf(o0[r] * inv);
      sa[base + 16 + (lane & 15)] = f2bf(o1[r] * inv);
    }
  }
}

// ---------------- elementwise producers (one-time conversions) ---------------
__global__ void k_add_bf(const float* __restrict__ a, const float* __restrict__ b,
                         unsigned short* __restrict__ o, int n) {
  int i = blockIdx.x * 256 + threadIdx.x;
  if (i < n) o[i] = f2bf(a[i] + b[i]);
}
__global__ void k_cvt(const float* __restrict__ a, unsigned short* __restrict__ o,
                      int n) {
  int i = blockIdx.x * 256 + threadIdx.x;
  if (i < n) o[i] = f2bf(a[i]);
}
// W (K,N) f32 -> Wt (N,K) bf16
__global__ void k_cvt_wt(const float* __restrict__ W,
                         unsigned short* __restrict__ Wt, int K, int N) {
  int i = blockIdx.x * 256 + threadIdx.x;
  if (i >= N * K) return;
  int n = i / K, k = i % K;
  Wt[i] = f2bf(W[(size_t)k * N + n]);
}
// vh (T*B, D) bf16 -> vtb (B*D, TP_) bf16, zero-padded past T
__global__ void k_vt(const unsigned short* __restrict__ vh,
                     unsigned short* __restrict__ vtb) {
  int i = blockIdx.x * 256 + threadIdx.x;
  if (i >= B_ * D_ * TP_) return;
  int t  = i % TP_;
  int bd = i / TP_;
  int b  = bd / D_, c = bd % D_;
  vtb[i] = (t < T_) ? vh[((size_t)t * B_ + b) * D_ + c] : (unsigned short)0;
}

// out = LN(x + y) * g + b ; one wave per 256-wide row; optional bf16 copy
__global__ __launch_bounds__(256) void k_ln_res(
    const float* __restrict__ x, const float* __restrict__ y,
    const float* __restrict__ gam, const float* __restrict__ bet,
    float* __restrict__ out, unsigned short* __restrict__ outb, int Mrows) {
  int row = blockIdx.x * 8 + (threadIdx.x >> 5);
  if (row >= Mrows) return;
  int lane = threadIdx.x & 31;
  const float* xr = x + (size_t)row * D_;
  const float* yr = y + (size_t)row * D_;
  float v[8]; float s = 0.f;
#pragma unroll
  for (int i = 0; i < 8; ++i) { v[i] = xr[lane + i * 32] + yr[lane + i * 32]; s += v[i]; }
#pragma unroll
  for (int sft = 1; sft < 32; sft <<= 1) s += __shfl_xor(s, sft, 32);
  float mean = s * (1.0f / 256.0f);
  float q = 0.f;
#pragma unroll
  for (int i = 0; i < 8; ++i) { float d = v[i] - mean; q += d * d; }
#pragma unroll
  for (int sft = 1; sft < 32; sft <<= 1) q += __shfl_xor(q, sft, 32);
  float rstd = rsqrtf(q * (1.0f / 256.0f) + 1e-5f);
#pragma unroll
  for (int i = 0; i < 8; ++i) {
    int c = lane + i * 32;
    float r = (v[i] - mean) * rstd * gam[c] + bet[c];
    out[(size_t)row * D_ + c] = r;
    if (outb) outb[(size_t)row * D_ + c] = f2bf(r);
  }
}

// in-place softmax over contiguous groups of 16
__global__ void k_softmax16(float* __restrict__ aw, int groups) {
  int gi = blockIdx.x * 256 + threadIdx.x;
  if (gi >= groups) return;
  float* p = aw + (size_t)gi * 16;
  float m = -3.0e38f;
#pragma unroll
  for (int i = 0; i < 16; ++i) m = fmaxf(m, p[i]);
  float s = 0.f;
#pragma unroll
  for (int i = 0; i < 16; ++i) { float e = __expf(p[i] - m); p[i] = e; s += e; }
  float inv = 1.0f / s;
#pragma unroll
  for (int i = 0; i < 16; ++i) p[i] *= inv;
}

// deformable bilinear sampling: one wave per (t,b,h), lane = channel; bf16 out
__global__ __launch_bounds__(256) void k_sample(
    const unsigned short* __restrict__ value,  // (LV*B, 256) bf16
    const float* __restrict__ off,             // (M1, 256)
    const float* __restrict__ aw,              // (M1, 128)
    const float* __restrict__ ref,             // (T, B, L, 4)
    unsigned short* __restrict__ ca) {         // (M1, 256) bf16
  int wid = blockIdx.x * 8 + (threadIdx.x >> 5);
  int row = wid >> 3, h = wid & 7;
  int lane = threadIdx.x & 31;
  int b = row % B_;
  const float* offr = off + (size_t)row * 256 + h * (L_ * P_ * 2);
  const float* awr  = aw  + (size_t)row * 128 + h * (L_ * P_);
  const float* refr = ref + (size_t)row * (L_ * 4);
  const int Hs[4] = {100, 50, 25, 13};
  const int St[4] = {0, 10000, 12500, 13125};
  float acc = 0.f;
#pragma unroll
  for (int l = 0; l < L_; ++l) {
    const int Hl = Hs[l], Wl = Hs[l], st = St[l];
    float cx = refr[l * 4 + 0], cy = refr[l * 4 + 1];
    float rw = refr[l * 4 + 2], rh = refr[l * 4 + 3];
#pragma unroll
    for (int p = 0; p < P_; ++p) {
      float ox = offr[(l * P_ + p) * 2 + 0], oy = offr[(l * P_ + p) * 2 + 1];
      float wgt = awr[l * P_ + p];
      float x = (cx + ox * 0.125f * rw) * Wl - 0.5f;  // off/P * 0.5
      float y = (cy + oy * 0.125f * rh) * Hl - 0.5f;
      float x0f = floorf(x), y0f = floorf(y);
      float fx = x - x0f, fy = y - y0f;
      int x0 = (int)x0f, y0 = (int)y0f;
      auto tap = [&](int yi, int xi) -> float {
        if (yi < 0 || yi >= Hl || xi < 0 || xi >= Wl) return 0.f;
        size_t vrow = (size_t)(st + yi * Wl + xi) * B_ + b;
        return bf2f(value[vrow * 256 + h * DH_ + lane]);
      };
      float s00 = tap(y0, x0),     s01 = tap(y0, x0 + 1);
      float s10 = tap(y0 + 1, x0), s11 = tap(y0 + 1, x0 + 1);
      acc += wgt * (s00 * (1 - fy) * (1 - fx) + s01 * (1 - fy) * fx +
                    s10 * fy * (1 - fx)       + s11 * fy * fx);
    }
  }
  ca[(size_t)row * 256 + h * DH_ + lane] = f2bf(acc);
}

// ---------------- host-side orchestration ------------------------------------
extern "C" void kernel_launch(void* const* d_in, const int* in_sizes, int n_in,
                              void* d_out, int out_size, void* d_ws, size_t ws_size,
                              hipStream_t stream) {
  (void)in_sizes; (void)n_in; (void)out_size; (void)ws_size;
  const float* tgt    = (const float*)d_in[0];
  const float* qpos   = (const float*)d_in[1];
  const float* ref    = (const float*)d_in[2];
  const float* memory = (const float*)d_in[3];
  const float* wq = (const float*)d_in[6];  const float* bq = (const float*)d_in[7];
  const float* wk = (const float*)d_in[8];  const float* bk = (const float*)d_in[9];
  const float* wv = (const float*)d_in[10]; const float* bv = (const float*)d_in[11];
  const float* wo = (const float*)d_in[12]; const float* bo = (const float*)d_in[13];
  const float* n1g = (const float*)d_in[14]; const float* n1b = (const float*)d_in[15];
  const float* n2g = (const float*)d_in[16]; const float* n2b = (const float*)d_in[17];
  const float* n3g = (const float*)d_in[18]; const float* n3b = (const float*)d_in[19];
  const float* offw = (const float*)d_in[20]; const float* offbias = (const float*)d_in[21];
  const float* aww  = (const float*)d_in[22]; const float* awbias  = (const float*)d_in[23];
  const float* valw = (const float*)d_in[24]; const float* valbias = (const float*)d_in[25];
  const float* opw  = (const float*)d_in[26]; const float* opbias  = (const float*)d_in[27];
  const float* l1w  = (const float*)d_in[28]; const float* l1b = (const float*)d_in[29];
  const float* l2w  = (const float*)d_in[30]; const float* l2b = (const float*)d_in[31];

  char* ws = (char*)d_ws;
  size_t o = 0;
  auto alloc = [&](size_t bytes) { size_t r = o; o += (bytes + 255) & ~(size_t)255; return r; };
  unsigned short* wq_t  = (unsigned short*)(ws + alloc(256 * 256 * 2));
  unsigned short* wk_t  = (unsigned short*)(ws + alloc(256 * 256 * 2));
  unsigned short* wv_t  = (unsigned short*)(ws + alloc(256 * 256 * 2));
  unsigned short* wo_t  = (unsigned short*)(ws + alloc(256 * 256 * 2));
  unsigned short* val_t = (unsigned short*)(ws + alloc(256 * 256 * 2));
  unsigned short* op_t  = (unsigned short*)(ws + alloc(256 * 256 * 2));
  unsigned short* off_t = (unsigned short*)(ws + alloc(256 * 256 * 2));
  unsigned short* aw_t  = (unsigned short*)(ws + alloc(128 * 256 * 2));
  unsigned short* l1_t  = (unsigned short*)(ws + alloc(1024 * 256 * 2));
  unsigned short* l2_t  = (unsigned short*)(ws + alloc(256 * 1024 * 2));
  // bf16 activations
  unsigned short* qbuf_bf = (unsigned short*)(ws + alloc((size_t)M1_ * 256 * 2)); // later qc
  unsigned short* tgt_bf  = (unsigned short*)(ws + alloc((size_t)M1_ * 256 * 2));
  unsigned short* qh_bf   = (unsigned short*)(ws + alloc((size_t)M1_ * 256 * 2));
  unsigned short* kh_bf   = (unsigned short*)(ws + alloc((size_t)M1_ * 256 * 2));
  unsigned short* vh_bf   = (unsigned short*)(ws + alloc((size_t)M1_ * 256 * 2));
  unsigned short* sa_bf   = (unsigned short*)(ws + alloc((size_t)M1_ * 256 * 2));
  unsigned short* ca_bf   = (unsigned short*)(ws + alloc((size_t)M1_ * 256 * 2));
  unsigned short* t3_bf   = (unsigned short*)(ws + alloc((size_t)M1_ * 256 * 2));
  unsigned short* vtb     = (unsigned short*)(ws + alloc((size_t)B_ * D_ * TP_ * 2));
  unsigned short* ff1_bf  = (unsigned short*)(ws + alloc((size_t)M1_ * 1024 * 2));
  // f32 buffers
  float* proj = (float*)(ws + alloc((size_t)M1_ * 256 * 4));
  float* tgt2 = (float*)(ws + alloc((size_t)M1_ * 256 * 4));
  float* tgt3 = (float*)(ws + alloc((size_t)M1_ * 256 * 4));
  float* offb = (float*)(ws + alloc((size_t)M1_ * 256 * 4));
  float* awb  = (float*)(ws + alloc((size_t)M1_ * 128 * 4));
  // big bf16 buffers
  unsigned short* mem_bf = (unsigned short*)(ws + alloc((size_t)MV_ * 256 * 2));
  unsigned short* valbf  = (unsigned short*)(ws + alloc((size_t)MV_ * 256 * 2));
  unsigned short* qc_bf = qbuf_bf;  // reuse: qbuf done before qc needed

  auto cvtw = [&](const float* W, unsigned short* Wt, int K, int N) {
    int n = N * K;
    k_cvt_wt<<<(n + 255) / 256, 256, 0, stream>>>(W, Wt, K, N);
  };
  auto gemm = [&](const unsigned short* A, int lda, const unsigned short* Wt,
                  const float* bias, float* Cf, unsigned short* Cb, int ldc,
                  int M, int N, int K, int relu) {
    dim3 grid((M + 127) / 128, (N + 127) / 128);
    k_gemm<<<grid, 256, 0, stream>>>(A, lda, Wt, bias, Cf, Cb, ldc, M, N, K, relu);
  };

  // 1. weight conversion (transpose to (N,K) bf16)
  cvtw(wq, wq_t, 256, 256);   cvtw(wk, wk_t, 256, 256);
  cvtw(wv, wv_t, 256, 256);   cvtw(wo, wo_t, 256, 256);
  cvtw(valw, val_t, 256, 256); cvtw(opw, op_t, 256, 256);
  cvtw(offw, off_t, 256, 256); cvtw(aww, aw_t, 256, 128);
  cvtw(l1w, l1_t, 256, 1024);  cvtw(l2w, l2_t, 1024, 256);

  const int nElem = M1_ * 256;
  const int nMem  = MV_ * 256;
  // 2. bf16 inputs: q = tgt + qpos; tgt copy; memory copy
  k_add_bf<<<(nElem + 255) / 256, 256, 0, stream>>>(tgt, qpos, qbuf_bf, nElem);
  k_cvt<<<(nElem + 255) / 256, 256, 0, stream>>>(tgt, tgt_bf, nElem);
  k_cvt<<<(nMem + 255) / 256, 256, 0, stream>>>(memory, mem_bf, nMem);
  // 3. projections (bf16 out)
  gemm(qbuf_bf, 256, wq_t, bq, nullptr, qh_bf, 256, M1_, 256, 256, 0);
  gemm(qbuf_bf, 256, wk_t, bk, nullptr, kh_bf, 256, M1_, 256, 256, 0);
  gemm(tgt_bf,  256, wv_t, bv, nullptr, vh_bf, 256, M1_, 256, 256, 0);
  // 3b. transpose V for b128 attention loads
  k_vt<<<(B_ * D_ * TP_ + 255) / 256, 256, 0, stream>>>(vh_bf, vtb);
  // 4. self-attention (bf16 out)
  k_attn<<<dim3(B_ * H_, 8), 256, 0, stream>>>(qh_bf, kh_bf, vtb, sa_bf);
  // 5. output projection + residual + LN2
  gemm(sa_bf, 256, wo_t, bo, proj, nullptr, 256, M1_, 256, 256, 0);
  k_ln_res<<<M1_ / 8, 256, 0, stream>>>(proj, tgt, n2g, n2b, tgt2, nullptr, M1_);
  // 6. cross-attn query (bf16)
  k_add_bf<<<(nElem + 255) / 256, 256, 0, stream>>>(tgt2, qpos, qc_bf, nElem);
  // 7. value projection (bf16 out) -- the big GEMM
  gemm(mem_bf, 256, val_t, valbias, nullptr, valbf, 256, MV_, 256, 256, 0);
  // 8. sampling offsets (f32 out)
  gemm(qc_bf, 256, off_t, offbias, offb, nullptr, 256, M1_, 256, 256, 0);
  // 9. attention weights + group softmax
  gemm(qc_bf, 256, aw_t, awbias, awb, nullptr, 128, M1_, 128, 256, 0);
  k_softmax16<<<(M1_ * 8 + 255) / 256, 256, 0, stream>>>(awb, M1_ * 8);
  // 10. deformable sampling (bf16 out)
  k_sample<<<M1_, 256, 0, stream>>>(valbf, offb, awb, ref, ca_bf);
  // 11. output projection + residual + LN1 (f32 + bf16 for FFN input)
  gemm(ca_bf, 256, op_t, opbias, proj, nullptr, 256, M1_, 256, 256, 0);
  k_ln_res<<<M1_ / 8, 256, 0, stream>>>(proj, tgt2, n1g, n1b, tgt3, t3_bf, M1_);
  // 12-13. FFN + residual + LN3
  gemm(t3_bf, 256, l1_t, l1b, nullptr, ff1_bf, 1024, M1_, 1024, 256, 1);
  gemm(ff1_bf, 1024, l2_t, l2b, proj, nullptr, 256, M1_, 256, 1024, 0);
  k_ln_res<<<M1_ / 8, 256, 0, stream>>>(proj, tgt3, n3g, n3b, (float*)d_out, nullptr, M1_);
}
